// Transformer_11510512353703
// MI455X (gfx1250) — compile-verified, matrix operand order checked by reading, not run on hardware
//
#include <hip/hip_runtime.h>
#include <hip/hip_bf16.h>
#include <stdint.h>

typedef __attribute__((ext_vector_type(16))) _Float16 v16h;
typedef __attribute__((ext_vector_type(8)))  _Float16 v8h;
typedef __attribute__((ext_vector_type(8)))  float    v8f;
typedef __attribute__((ext_vector_type(4)))  float    v4f;

#define T_TOK   5888   // 128 * 46
#define C_DIM   512
#define CI_DIM  256
#define N_BATCH 128
#define BLKSZ   46

// GEMM macro-tile: 256 threads = 8 waves in a 2x4 grid; each wave owns a
// 16x32 output strip (two 16x16 WMMA tiles sharing one A fragment).
#define BM      32
#define BN      128
#define KSTEP   64     // two wmma k-steps per LDS stage
#define LDH_P   72     // padded LDS row stride in halves  (144 B, 16B-aligned)
#define LDF_P   68     // padded LDS row stride in floats  (272 B, 16B-aligned)

// ---------------------------------------------------------------------------
// gfx1250 async-copy path (ASYNCcnt): global -> LDS without VGPR staging.
// LDS destination offset = low 32 bits of the flat shared-memory address.
// ---------------------------------------------------------------------------
__device__ __forceinline__ void async_b128(uint32_t lds_byte_off, uint64_t gaddr) {
  asm volatile("global_load_async_to_lds_b128 %0, %1, off"
               :: "v"(lds_byte_off), "v"(gaddr) : "memory");
}
__device__ __forceinline__ void wait_async0() {
  asm volatile("s_wait_asynccnt 0x0" ::: "memory");
}
__device__ __forceinline__ uint32_t lds_off(const void* p) {
  return (uint32_t)(uintptr_t)p;
}

// ---------------------------------------------------------------------------
// Cooperative tile loaders (whole 256-thread block), one b128 per issue.
// ---------------------------------------------------------------------------
template <typename AT, int PAD>
__device__ __forceinline__ void load_tileA(const AT* __restrict__ A, int lda,
                                           int mBlock, int k0,
                                           AT (*sA)[PAD], int tid) {
  constexpr int EPC = 16 / sizeof(AT);   // elements per 16B chunk: 8 h / 4 f
  constexpr int CPR = 64 / EPC;          // chunks per 64-element row: 8 / 16
  constexpr int ITER = (BM * CPR) / 256; // 1 (f16) / 2 (f32)
#pragma unroll
  for (int i = 0; i < ITER; ++i) {
    const int id = tid + i * 256;
    const int r  = id / CPR;
    const int cc = (id % CPR) * EPC;
    async_b128(lds_off(&sA[r][cc]),
               (uint64_t)(uintptr_t)(A + (size_t)(mBlock + r) * lda + k0 + cc));
  }
}
__device__ __forceinline__ void load_tileB(const _Float16* __restrict__ B, int ldb,
                                           int nBlock, int k0,
                                           _Float16 (*sB)[LDH_P], int tid) {
#pragma unroll
  for (int i = 0; i < 4; ++i) {          // 128 rows x 64 halves = 1024 chunks
    const int id = tid + i * 256;
    const int r = id >> 3, cc = (id & 7) * 8;
    async_b128(lds_off(&sB[r][cc]),
               (uint64_t)(uintptr_t)(B + (size_t)(nBlock + r) * ldb + k0 + cc));
  }
}

// ---------------------------------------------------------------------------
// Fragment builders from LDS (layout per CDNA5 ISA §7.12.2):
// lane L holds row (L&15); its 16 halves are the two contiguous 8-half runs
// at K offsets (L>>4)*8 and 16+(L>>4)*8 within the 32-wide k-step.
// ---------------------------------------------------------------------------
__device__ __forceinline__ v16h make_frag(const _Float16 (*sm)[LDH_P],
                                          int rBase, int kk, int lane) {
  const int m = lane & 15, h = lane >> 4;
  const _Float16* row = &sm[rBase + m][kk * 32 + h * 8];
  v8h lo = *(const v8h*)row;
  v8h hi = *(const v8h*)(row + 16);
  v16h r;
#pragma unroll
  for (int i = 0; i < 8; ++i) { r[i] = lo[i]; r[i + 8] = hi[i]; }
  return r;
}
__device__ __forceinline__ v16h make_frag(const float (*sm)[LDF_P],
                                          int rBase, int kk, int lane) {
  const int m = lane & 15, h = lane >> 4;
  const float* row = &sm[rBase + m][kk * 32 + h * 8];
  v4f a0 = *(const v4f*)(row);
  v4f a1 = *(const v4f*)(row + 4);
  v4f b0 = *(const v4f*)(row + 16);
  v4f b1 = *(const v4f*)(row + 20);
  v16h r;
#pragma unroll
  for (int i = 0; i < 4; ++i) {
    r[i]      = (_Float16)a0[i];
    r[i + 4]  = (_Float16)a1[i];
    r[i + 8]  = (_Float16)b0[i];
    r[i + 12] = (_Float16)b1[i];
  }
  return r;
}

// ---------------------------------------------------------------------------
// WMMA GEMM: D = A(MxK) @ Bt(NxK)^T  (+ epilogue). Async-LDS double-buffered.
//   MODE 0: f16 out, row-major, + bias (bias may be null)
//   MODE 1: f16 out, TRANSPOSED (D[col*ldd + row]), + bias   (g_x^T)
//   MODE 2: f32 out, block-diagonal mask to -1000             (attn logits)
//   MODE 3: f32 out, + bias + ori_feature^T add               (final W_y)
// M % 32 == 0, N % 128 == 0, K % 64 == 0 for all launches here.
// ---------------------------------------------------------------------------
template <typename AT, int MODE>
__global__ void __launch_bounds__(256)
wmma_gemm(const AT* __restrict__ A, int lda,
          const _Float16* __restrict__ B, int ldb,
          void* __restrict__ Dptr, int ldd,
          const float* __restrict__ bias,
          int K, const float* __restrict__ ori) {
  constexpr int LDA_PAD = (sizeof(AT) == 4) ? LDF_P : LDH_P;
  __shared__ alignas(16) AT       sA[2][BM][LDA_PAD];
  __shared__ alignas(16) _Float16 sB[2][BN][LDH_P];

  const int tid    = threadIdx.x;
  const int lane   = tid & 31;
  const int wave   = tid >> 5;
  const int mSub   = (wave >> 2) * 16;      // 2 wave-rows
  const int nSub   = (wave & 3) * 32;       // 4 wave-cols, 32 wide each
  const int mBlock = blockIdx.y * BM;
  const int nBlock = blockIdx.x * BN;

  v8f acc[2] = {{}, {}};
  const int nStages = K / KSTEP;

  load_tileA<AT, LDA_PAD>(A, lda, mBlock, 0, sA[0], tid);
  load_tileB(B, ldb, nBlock, 0, sB[0], tid);
  wait_async0();
  __syncthreads();

  for (int s = 0; s < nStages; ++s) {
    const int cur = s & 1;
    if (s + 1 < nStages) {                  // prefetch next stage, other buffer
      const int nxt = (s + 1) & 1;
      const int k0 = (s + 1) * KSTEP;
      load_tileA<AT, LDA_PAD>(A, lda, mBlock, k0, sA[nxt], tid);
      load_tileB(B, ldb, nBlock, k0, sB[nxt], tid);
    }
#pragma unroll
    for (int kk = 0; kk < 2; ++kk) {
      v16h a  = make_frag(sA[cur], mSub, kk, lane);
      v16h b0 = make_frag(sB[cur], nSub, kk, lane);
      v16h b1 = make_frag(sB[cur], nSub + 16, kk, lane);
      acc[0] = __builtin_amdgcn_wmma_f32_16x16x32_f16(false, a, false, b0,
                                                      (short)0, acc[0], false, false);
      acc[1] = __builtin_amdgcn_wmma_f32_16x16x32_f16(false, a, false, b1,
                                                      (short)0, acc[1], false, false);
    }
    wait_async0();     // next stage landed in LDS
    __syncthreads();   // ...and everyone finished reading the current buffer
  }

  // D layout: VGPR r, lanes 0-15 -> M=r, lanes 16-31 -> M=8+r; N = lane&15
  const int n  = lane & 15;
  const int mh = lane >> 4;
#pragma unroll
  for (int t = 0; t < 2; ++t) {
    const int col = nBlock + nSub + t * 16 + n;
    const float bv = (bias != nullptr) ? bias[col] : 0.0f;
#pragma unroll
    for (int r = 0; r < 8; ++r) {
      const int row = mBlock + mSub + mh * 8 + r;
      float v = acc[t][r] + bv;
      if constexpr (MODE == 0) {
        ((_Float16*)Dptr)[(size_t)row * ldd + col] = (_Float16)v;
      } else if constexpr (MODE == 1) {
        ((_Float16*)Dptr)[(size_t)col * ldd + row] = (_Float16)v;
      } else if constexpr (MODE == 2) {
        if ((row / BLKSZ) == (col / BLKSZ)) v = -1000.0f;
        ((float*)Dptr)[(size_t)row * ldd + col] = v;
      } else {
        const int nb = row / BLKSZ;
        const int j  = row - nb * BLKSZ;
        v += ori[(size_t)nb * (C_DIM * BLKSZ) + (size_t)col * BLKSZ + j];
        ((float*)Dptr)[(size_t)row * ldd + col] = v;
      }
    }
  }
}

// ---------------------------------------------------------------------------
// BatchNorm (inference) + ReLU + (N,C,46)->(T,C) transpose, output f16.
// ---------------------------------------------------------------------------
__global__ void bn_relu_transpose(const float* __restrict__ ori,
                                  const float* __restrict__ gamma,
                                  const float* __restrict__ beta,
                                  const float* __restrict__ mean,
                                  const float* __restrict__ var,
                                  _Float16* __restrict__ feat) {
  const int idx = blockIdx.x * blockDim.x + threadIdx.x;  // over T*C
  if (idx >= T_TOK * C_DIM) return;
  const int c = idx & (C_DIM - 1);
  const int t = idx >> 9;
  const int nb = t / BLKSZ;
  const int j  = t - nb * BLKSZ;
  const float inv = gamma[c] * rsqrtf(var[c] + 1e-5f);
  const float sh  = beta[c] - mean[c] * inv;
  const float x   = ori[(size_t)nb * (C_DIM * BLKSZ) + (size_t)c * BLKSZ + j];
  feat[idx] = (_Float16)fmaxf(x * inv + sh, 0.0f);
}

__global__ void cvt_f32_to_f16(const float* __restrict__ src,
                               _Float16* __restrict__ dst, int n) {
  const int i = blockIdx.x * blockDim.x + threadIdx.x;
  if (i < n) dst[i] = (_Float16)src[i];
}

// ---------------------------------------------------------------------------
// Row softmax, in place over the (T x T) fp32 attention matrix in d_out.
// ---------------------------------------------------------------------------
__global__ void softmax_rows(float* __restrict__ P) {
  const int row = blockIdx.x;
  float* p = P + (size_t)row * T_TOK;
  __shared__ float red[256];
  const int tid = threadIdx.x;

  float m = -3.4e38f;
  for (int i = tid; i < T_TOK; i += 256) m = fmaxf(m, p[i]);
  red[tid] = m; __syncthreads();
  for (int s = 128; s > 0; s >>= 1) {
    if (tid < s) red[tid] = fmaxf(red[tid], red[tid + s]);
    __syncthreads();
  }
  m = red[0]; __syncthreads();

  float sum = 0.0f;
  for (int i = tid; i < T_TOK; i += 256) {
    const float e = __expf(p[i] - m);
    p[i] = e;                       // store exp, rescale in pass 3
    sum += e;
  }
  red[tid] = sum; __syncthreads();
  for (int s = 128; s > 0; s >>= 1) {
    if (tid < s) red[tid] += red[tid + s];
    __syncthreads();
  }
  const float inv = 1.0f / red[0];
  for (int i = tid; i < T_TOK; i += 256) p[i] *= inv;
}

// ---------------------------------------------------------------------------
extern "C" void kernel_launch(void* const* d_in, const int* in_sizes, int n_in,
                              void* d_out, int out_size, void* d_ws, size_t ws_size,
                              hipStream_t stream) {
  (void)in_sizes; (void)n_in; (void)out_size; (void)ws_size;

  const float* ori     = (const float*)d_in[0];
  const float* gamma   = (const float*)d_in[1];
  const float* beta    = (const float*)d_in[2];
  const float* mean    = (const float*)d_in[3];
  const float* var     = (const float*)d_in[4];
  const float* theta_w = (const float*)d_in[5];
  const float* theta_b = (const float*)d_in[6];
  const float* phi_w   = (const float*)d_in[7];
  const float* phi_b   = (const float*)d_in[8];
  const float* g_w     = (const float*)d_in[9];
  const float* g_b     = (const float*)d_in[10];
  const float* W_w     = (const float*)d_in[11];
  const float* W_b     = (const float*)d_in[12];

  float* att_fea = (float*)d_out;
  float* fdiv    = att_fea + (size_t)N_BATCH * BLKSZ * C_DIM;

  size_t off = 0;
  auto take = [&](size_t bytes) -> void* {
    void* p = (char*)d_ws + off;
    off = (off + bytes + 255) & ~(size_t)255;
    return p;
  };
  _Float16* feat16 = (_Float16*)take((size_t)T_TOK * C_DIM * 2);
  _Float16* thW16  = (_Float16*)take((size_t)CI_DIM * C_DIM * 2);
  _Float16* phW16  = (_Float16*)take((size_t)CI_DIM * C_DIM * 2);
  _Float16* gW16   = (_Float16*)take((size_t)CI_DIM * C_DIM * 2);
  _Float16* WW16   = (_Float16*)take((size_t)C_DIM * CI_DIM * 2);
  _Float16* xth16  = (_Float16*)take((size_t)T_TOK * CI_DIM * 2);
  _Float16* xph16  = (_Float16*)take((size_t)T_TOK * CI_DIM * 2);
  _Float16* gxT16  = (_Float16*)take((size_t)CI_DIM * T_TOK * 2);  // transposed
  _Float16* y16    = (_Float16*)take((size_t)T_TOK * CI_DIM * 2);

  const int WCNT = CI_DIM * C_DIM;
  cvt_f32_to_f16<<<(WCNT + 255) / 256, 256, 0, stream>>>(theta_w, thW16, WCNT);
  cvt_f32_to_f16<<<(WCNT + 255) / 256, 256, 0, stream>>>(phi_w,   phW16, WCNT);
  cvt_f32_to_f16<<<(WCNT + 255) / 256, 256, 0, stream>>>(g_w,     gW16,  WCNT);
  cvt_f32_to_f16<<<(WCNT + 255) / 256, 256, 0, stream>>>(W_w,     WW16,  WCNT);

  bn_relu_transpose<<<(T_TOK * C_DIM + 255) / 256, 256, 0, stream>>>(
      ori, gamma, beta, mean, var, feat16);

  const dim3 gProj(CI_DIM / BN, T_TOK / BM);   // (2, 184)
  // x_theta = feat @ theta_w^T + theta_b
  wmma_gemm<_Float16, 0><<<gProj, 256, 0, stream>>>(
      feat16, C_DIM, thW16, C_DIM, xth16, CI_DIM, theta_b, C_DIM, nullptr);
  // x_phi = feat @ phi_w^T + phi_b
  wmma_gemm<_Float16, 0><<<gProj, 256, 0, stream>>>(
      feat16, C_DIM, phW16, C_DIM, xph16, CI_DIM, phi_b, C_DIM, nullptr);
  // g_x^T = (feat @ g_w^T + g_b)^T
  wmma_gemm<_Float16, 1><<<gProj, 256, 0, stream>>>(
      feat16, C_DIM, gW16, C_DIM, gxT16, T_TOK, g_b, C_DIM, nullptr);

  // attn logits = x_theta @ x_phi^T, block-diag masked, fp32 into d_out
  wmma_gemm<_Float16, 2><<<dim3(T_TOK / BN, T_TOK / BM), 256, 0, stream>>>(
      xth16, CI_DIM, xph16, CI_DIM, fdiv, T_TOK, nullptr, CI_DIM, nullptr);

  softmax_rows<<<T_TOK, 256, 0, stream>>>(fdiv);

  // y = P @ g_x  (A fp32, staged to LDS and converted on fragment load)
  wmma_gemm<float, 0><<<gProj, 256, 0, stream>>>(
      fdiv, T_TOK, gxT16, T_TOK, y16, CI_DIM, nullptr, T_TOK, nullptr);

  // att_fea = y @ W_w^T + W_b + ori^T, fp32 into d_out
  wmma_gemm<_Float16, 3><<<dim3(C_DIM / BN, T_TOK / BM), 256, 0, stream>>>(
      y16, CI_DIM, WW16, CI_DIM, att_fea, C_DIM, W_b, CI_DIM, ori);
}
